// Attention_27633819583041
// MI455X (gfx1250) — matrix operand order repaired and checked by reading, not compiled, not run on hardware
//
#include <hip/hip_runtime.h>
#include <hip/hip_bf16.h>

typedef unsigned short u16;
typedef __attribute__((ext_vector_type(16))) __bf16 bf16x16;
typedef __attribute__((ext_vector_type(8)))  float  f32x8;

union BOp {                   // one WMMA bf16 operand: 8 VGPRs / 16 bf16 / 32 bytes
    bf16x16 v;
    u16     s[16];
    uint4   q[2];
};

#define NB  8
#define NS  1024
#define ND  1024
#define NH  8
#define HD  128
#define PREFIX_LEN 77
#define MASKED_BIAS -10000.0f
#define SCALE 0.08838834764831845f   // 1/sqrt(128)

// hardware f32->bf16 (v_cvt_*_bf16_f32, RNE) instead of manual bit-twiddling
static __device__ __forceinline__ u16 f2bf(float f) {
    union { __bf16 b; u16 u; } c;
    c.b = (__bf16)f;
    return c.u;
}

// ---- async global->LDS copy (ASYNCcnt-tracked), 16 bytes per lane ----------
// Builtin parameter type (from hipcc diagnostic): non-const addrspace(1)
// pointer to `int __attribute__((__vector_size__(16)))`.
typedef int v4i_async __attribute__((__vector_size__(16)));

static __device__ __forceinline__ void async_copy_b128(const void* gsrc, void* lds_dst) {
#if __has_builtin(__builtin_amdgcn_global_load_async_to_lds_b128)
    __builtin_amdgcn_global_load_async_to_lds_b128(
        (__attribute__((address_space(1))) v4i_async*)(void*)gsrc,
        (__attribute__((address_space(3))) v4i_async*)lds_dst, 0, 0);
#else
    // VDST = LDS byte address (low 32 bits of generic LDS pointer), VADDR = 64b global
    unsigned lds_off = (unsigned)(unsigned long long)lds_dst;
    asm volatile("global_load_async_to_lds_b128 %0, %1, off"
                 :: "v"(lds_off), "v"(gsrc) : "memory");
#endif
}

static __device__ __forceinline__ void wait_async0() {
#if __has_builtin(__builtin_amdgcn_s_wait_asynccnt)
    __builtin_amdgcn_s_wait_asynccnt(0);
#else
    asm volatile("s_wait_asynccnt 0" ::: "memory");
#endif
}

// ---------------------------------------------------------------------------
// Kernel 1: QKV GEMM.  x[8192,1024] (f32) @ W_attn[1024,3072] + b_attn.
// Output scattered as bf16 into Q/K/V in [B,H,S,hd] layout.
// Block: 256 thr (8 waves), tile 128(M) x 64(N), K-step 32.
// ---------------------------------------------------------------------------
__global__ __launch_bounds__(256) void k_qkv_gemm(
    const float* __restrict__ x, const float* __restrict__ W,
    const float* __restrict__ bias,
    u16* __restrict__ Q, u16* __restrict__ K, u16* __restrict__ V)
{
    __shared__ __align__(16) u16 As[128 * 40];   // [row][k], pitch 40 (80B, 16B-aligned)
    __shared__ __align__(16) u16 Bs[64 * 40];    // [col][k] (transposed W tile)

    const int t    = threadIdx.x;
    const int wave = t >> 5;
    const int lane = t & 31;
    const int lo   = lane & 15;
    const int hi   = lane >> 4;
    const int Nbase = blockIdx.x * 64;
    const int Mbase = blockIdx.y * 128;

    f32x8 acc[4] = {};

    for (int k0 = 0; k0 < ND; k0 += 32) {
        // stage A tile: convert 16 f32 -> bf16 in regs, two b128 LDS stores
        {
            const int row = t >> 1, cs = (t & 1) * 16;
            const float* src = x + (size_t)(Mbase + row) * ND + k0 + cs;
            __builtin_prefetch(src + 32, 0, 0);            // next K tile -> global_prefetch
            BOp tmp;
            #pragma unroll
            for (int i = 0; i < 16; ++i) tmp.s[i] = f2bf(src[i]);
            *(uint4*)&As[row * 40 + cs]     = tmp.q[0];
            *(uint4*)&As[row * 40 + cs + 8] = tmp.q[1];
        }
        // stage B tile transposed: 2 k-rows x 4 cols per thread, packed b32 stores
        {
            const int kr = (t >> 4) * 2, cg = (t & 15) * 4;
            const float* s0 = W + (size_t)(k0 + kr) * 3072 + Nbase + cg;
            const float* s1 = s0 + 3072;
            __builtin_prefetch(s0 + (size_t)32 * 3072, 0, 0);
            #pragma unroll
            for (int i = 0; i < 4; ++i) {
                unsigned pk = (unsigned)f2bf(s0[i]) | ((unsigned)f2bf(s1[i]) << 16);
                *(unsigned*)&Bs[(cg + i) * 40 + kr] = pk;
            }
        }
        __syncthreads();

        // A operand: lane(lo,hi) -> row wave*16+lo, K = hi*16 + 0..15 (contiguous)
        BOp a;
        const u16* arow = &As[(wave * 16 + lo) * 40];
        a.q[0] = *(const uint4*)(arow + hi * 16);
        a.q[1] = *(const uint4*)(arow + hi * 16 + 8);

        #pragma unroll
        for (int nt = 0; nt < 4; ++nt) {
            // B operand: lane(lo,hi) -> col nt*16+lo, K = hi*16 + 0..15 (contiguous)
            BOp b;
            const u16* bcol = &Bs[(nt * 16 + lo) * 40 + hi * 16];
            b.q[0] = *(const uint4*)(bcol);
            b.q[1] = *(const uint4*)(bcol + 8);
            acc[nt] = __builtin_amdgcn_wmma_f32_16x16x32_bf16(
                false, a.v, false, b.v, (short)0, acc[nt], false, false);
        }
        __syncthreads();
    }

    // Scatter C tiles: VGPR r -> row (wave*16 + r + 8*hi), col nt*16+lo
    const int rowb = Mbase + wave * 16 + hi * 8;
    #pragma unroll
    for (int nt = 0; nt < 4; ++nt) {
        const int col   = Nbase + nt * 16 + lo;
        const float bv  = bias[col];
        const int which = col >> 10;
        const int dcol  = col & 1023;
        const int h     = dcol >> 7;
        const int e     = dcol & 127;
        u16* outp = (which == 0) ? Q : (which == 1) ? K : V;
        #pragma unroll
        for (int r = 0; r < 8; ++r) {
            const int row = rowb + r;
            const int bb  = row >> 10;
            const int ss  = row & 1023;
            outp[(((size_t)bb * NH + h) * NS + ss) * HD + e] = f2bf(acc[nt][r] + bv);
        }
    }
}

// ---------------------------------------------------------------------------
// Kernel 2: fused flash attention per (b,h).  Computes S^T = K.Q^T so the score
// C layout becomes the B operand layout of P^T after one shfl_xor(16); then
// accumulates O^T = V^T . P^T with V^T staged in LDS (shared by the 4 waves).
// Block: 128 thr (4 waves, 4 query tiles of 16), grid (S/64, H, B).
// ---------------------------------------------------------------------------
__global__ __launch_bounds__(128) void k_flash_attn(
    const u16* __restrict__ Qg, const u16* __restrict__ Kg,
    const u16* __restrict__ Vg, const float* __restrict__ amask,
    u16* __restrict__ Aout)
{
    __shared__ __align__(16) u16 Vt[128 * 40];   // V^T tile: [d][key], pitch 40

    const int t    = threadIdx.x;
    const int wave = t >> 5;
    const int lane = t & 31;
    const int lo   = lane & 15;
    const int hi   = lane >> 4;
    const int qb0  = blockIdx.x * 64;
    const int h    = blockIdx.y;
    const int b    = blockIdx.z;
    const int qbase = qb0 + wave * 16;
    const int qrow  = qbase + lo;          // this lane's query column

    const size_t bh = ((size_t)b * NH + h) * NS;
    const u16* Qp = Qg + bh * HD;
    const u16* Kp = Kg + bh * HD;
    const u16* Vp = Vg + bh * HD;

    // Q^T as 4 B operands (K=32 each over hd=128), kept in registers
    BOp Qb[4];
    #pragma unroll
    for (int c = 0; c < 4; ++c) {
        const u16* qsrc = Qp + (size_t)qrow * HD + c * 32 + hi * 8;
        Qb[c].q[0] = *(const uint4*)qsrc;
        Qb[c].q[1] = *(const uint4*)(qsrc + 16);
    }

    f32x8 O[8] = {};                       // O^T accumulators: 8 d-tiles of 16
    float m = -1e30f, l = 0.0f;

    int kend = qb0 + 64;                   // causal bound for this block
    if (kend < 96) kend = 96;              // cover the 77-wide prefix window
    if (kend > NS) kend = NS;

    for (int kb = 0; kb < kend; kb += 32) {
        // ---- stage V^T[d][key]: 2 key-rows x 16 d per thread, packed b32 stores ----
        {
            const int key = (t >> 3) * 2;          // 0..30 step 2
            const int dg  = (t & 7) * 16;          // 0..112 step 16
            const u16* v0 = Vp + (size_t)(kb + key) * HD + dg;
            const u16* v1 = v0 + HD;
            #pragma unroll
            for (int i = 0; i < 16; ++i) {
                unsigned pk = (unsigned)v0[i] | ((unsigned)v1[i] << 16);
                *(unsigned*)&Vt[(dg + i) * 40 + key] = pk;
            }
        }
        __syncthreads();

        // ---- scores S^T: A = K rows, B = Q^T;  two 16-key tiles ----
        f32x8 c0 = {}, c1 = {};
        const u16* k0p = Kp + (size_t)(kb + lo) * HD;
        const u16* k1p = Kp + (size_t)(kb + 16 + lo) * HD;
        #pragma unroll
        for (int c = 0; c < 4; ++c) {
            BOp a;
            a.q[0] = *(const uint4*)(k0p + c * 32 + hi * 8);
            a.q[1] = *(const uint4*)(k0p + c * 32 + 16 + hi * 8);
            c0 = __builtin_amdgcn_wmma_f32_16x16x32_bf16(
                false, a.v, false, Qb[c].v, (short)0, c0, false, false);
        }
        #pragma unroll
        for (int c = 0; c < 4; ++c) {
            BOp a;
            a.q[0] = *(const uint4*)(k1p + c * 32 + hi * 8);
            a.q[1] = *(const uint4*)(k1p + c * 32 + 16 + hi * 8);
            c1 = __builtin_amdgcn_wmma_f32_16x16x32_bf16(
                false, a.v, false, Qb[c].v, (short)0, c1, false, false);
        }

        // ---- mask + online softmax (per-lane query column) ----
        float s0[8], s1[8], p0[8], p1[8];
        float mx = -1e30f;
        #pragma unroll
        for (int r = 0; r < 8; ++r) {
            const int ka = kb + r + 8 * hi;        // tile0 key of VGPR r
            const int kc = ka + 16;                // tile1 key
            const bool a0 = (qrow < PREFIX_LEN) ? (ka < PREFIX_LEN) : (ka <= qrow);
            const bool a1 = (qrow < PREFIX_LEN) ? (kc < PREFIX_LEN) : (kc <= qrow);
            s0[r] = (a0 ? c0[r] * SCALE : MASKED_BIAS) + amask[(size_t)b * NS + ka];
            s1[r] = (a1 ? c1[r] * SCALE : MASKED_BIAS) + amask[(size_t)b * NS + kc];
            mx = fmaxf(mx, fmaxf(s0[r], s1[r]));
        }
        mx = fmaxf(mx, __shfl_xor(mx, 16, 32));    // combine the two hi-halves
        const float mn   = fmaxf(m, mx);
        const float corr = __expf(m - mn);
        #pragma unroll
        for (int dt = 0; dt < 8; ++dt)
            #pragma unroll
            for (int j = 0; j < 8; ++j) O[dt][j] *= corr;
        float ls = 0.0f;
        #pragma unroll
        for (int r = 0; r < 8; ++r) {
            p0[r] = __expf(s0[r] - mn);
            p1[r] = __expf(s1[r] - mn);
            ls += p0[r] + p1[r];
        }
        ls += __shfl_xor(ls, 16, 32);
        l = l * corr + ls;
        m = mn;

        // ---- P^T as WMMA B operand (K = 32 keys): one xor-16 exchange ----
        BOp Pb;
        #pragma unroll
        for (int r = 0; r < 8; ++r) {
            const float x0 = __shfl_xor(p0[r], 16, 32);
            const float x1 = __shfl_xor(p1[r], 16, 32);
            const float first  = hi ? x1    : p0[r];   // keys hi*16 + r
            const float second = hi ? p1[r] : x0;      // keys hi*16 + 8 + r
            Pb.s[r]     = f2bf(first);
            Pb.s[8 + r] = f2bf(second);
        }

        // ---- O^T += V^T . P^T  (A rows = hd, from LDS) ----
        #pragma unroll
        for (int dt = 0; dt < 8; ++dt) {
            BOp a;
            const u16* vrow = &Vt[(dt * 16 + lo) * 40];
            a.q[0] = *(const uint4*)(vrow + hi * 16);
            a.q[1] = *(const uint4*)(vrow + hi * 16 + 8);
            O[dt] = __builtin_amdgcn_wmma_f32_16x16x32_bf16(
                false, a.v, false, Pb.v, (short)0, O[dt], false, false);
        }
        __syncthreads();
    }

    // ---- normalize and store merged-head bf16 activations (b128 stores) ----
    const float invl = 1.0f / l;
    u16* Ap = Aout + ((size_t)b * NS + qrow) * ND + h * HD;
    #pragma unroll
    for (int dt = 0; dt < 8; ++dt) {
        BOp o;
        #pragma unroll
        for (int r = 0; r < 8; ++r) o.s[r] = f2bf(O[dt][r] * invl);
        *(uint4*)(Ap + dt * 16 + 8 * hi) = o.q[0];   // contiguous 8-run per d-tile
    }
}

// ---------------------------------------------------------------------------
// Kernel 3: output projection.  A(bf16)[8192,1024] @ W_proj[1024,1024] + b_proj
// -> f32 out.  A tile staged via GLOBAL_LOAD_ASYNC_TO_LDS (raw bf16 copy).
// ---------------------------------------------------------------------------
__global__ __launch_bounds__(256) void k_proj_gemm(
    const u16* __restrict__ A, const float* __restrict__ W,
    const float* __restrict__ bias, float* __restrict__ out)
{
    __shared__ __align__(16) u16 As[128 * 40];
    __shared__ __align__(16) u16 Bs[64 * 40];

    const int t    = threadIdx.x;
    const int wave = t >> 5;
    const int lane = t & 31;
    const int lo   = lane & 15;
    const int hi   = lane >> 4;
    const int Nbase = blockIdx.x * 64;
    const int Mbase = blockIdx.y * 128;

    f32x8 acc[4] = {};

    for (int k0 = 0; k0 < ND; k0 += 32) {
        {   // A tile already bf16: async DMA straight into LDS (2 x b128 / thread)
            const int row = t >> 1, cs = (t & 1) * 16;
            const u16* src = A + (size_t)(Mbase + row) * ND + k0 + cs;
            u16* dst = &As[row * 40 + cs];
            async_copy_b128(src, dst);
            async_copy_b128(src + 8, dst + 8);
        }
        {   // W tile transposed + converted: 2 k-rows x 4 cols, packed b32 stores
            const int kr = (t >> 4) * 2, cg = (t & 15) * 4;
            const float* s0 = W + (size_t)(k0 + kr) * ND + Nbase + cg;
            const float* s1 = s0 + ND;
            __builtin_prefetch(s0 + (size_t)32 * ND, 0, 0);
            #pragma unroll
            for (int i = 0; i < 4; ++i) {
                unsigned pk = (unsigned)f2bf(s0[i]) | ((unsigned)f2bf(s1[i]) << 16);
                *(unsigned*)&Bs[(cg + i) * 40 + kr] = pk;
            }
        }
        wait_async0();          // this wave's async LDS writes landed
        __syncthreads();        // all waves' writes visible

        BOp a;
        const u16* arow = &As[(wave * 16 + lo) * 40];
        a.q[0] = *(const uint4*)(arow + hi * 16);
        a.q[1] = *(const uint4*)(arow + hi * 16 + 8);

        #pragma unroll
        for (int nt = 0; nt < 4; ++nt) {
            BOp bop;
            const u16* bcol = &Bs[(nt * 16 + lo) * 40 + hi * 16];
            bop.q[0] = *(const uint4*)(bcol);
            bop.q[1] = *(const uint4*)(bcol + 8);
            acc[nt] = __builtin_amdgcn_wmma_f32_16x16x32_bf16(
                false, a.v, false, bop.v, (short)0, acc[nt], false, false);
        }
        __syncthreads();
    }

    const int rowb = Mbase + wave * 16 + hi * 8;
    #pragma unroll
    for (int nt = 0; nt < 4; ++nt) {
        const int col  = Nbase + nt * 16 + lo;
        const float bv = bias[col];
        #pragma unroll
        for (int r = 0; r < 8; ++r)
            out[(size_t)(rowb + r) * ND + col] = acc[nt][r] + bv;
    }
}

// ---------------------------------------------------------------------------
extern "C" void kernel_launch(void* const* d_in, const int* in_sizes, int n_in,
                              void* d_out, int out_size, void* d_ws, size_t ws_size,
                              hipStream_t stream)
{
    const float* x      = (const float*)d_in[0];
    const float* amask  = (const float*)d_in[1];
    const float* W_attn = (const float*)d_in[2];
    const float* b_attn = (const float*)d_in[3];
    const float* W_proj = (const float*)d_in[4];
    const float* b_proj = (const float*)d_in[5];
    float* out = (float*)d_out;

    const size_t tensor_elems = (size_t)NB * NH * NS * HD;   // 8,388,608 bf16 each
    u16* Q  = (u16*)d_ws;
    u16* K  = Q + tensor_elems;
    u16* V  = K + tensor_elems;
    u16* Aw = V + tensor_elems;                              // [B,S,D] bf16

    // 1) QKV projection -> bf16 Q/K/V in [B,H,S,hd]
    k_qkv_gemm<<<dim3(3072 / 64, 8192 / 128), 256, 0, stream>>>(
        x, W_attn, b_attn, Q, K, V);

    // 2) fused prefix-causal flash attention -> bf16 merged-head activations
    k_flash_attn<<<dim3(NS / 64, NH, NB), 128, 0, stream>>>(
        Q, K, V, amask, Aw);

    // 3) output projection -> f32
    k_proj_gemm<<<dim3(ND / 64, 8192 / 128), 256, 0, stream>>>(
        Aw, W_proj, b_proj, out);
}
